// PropagationBlock_11819749998953
// MI455X (gfx1250) — matrix operand order repaired and checked by reading, not compiled, not run on hardware
//
#include <hip/hip_runtime.h>
#include <math.h>

typedef __attribute__((ext_vector_type(16))) _Float16 v16h;
typedef __attribute__((ext_vector_type(8)))  _Float16 v8h;
typedef __attribute__((ext_vector_type(8)))  float    v8f;
typedef __attribute__((ext_vector_type(4)))  float    f32x4;

#define WMMA_F16(A,B,C) __builtin_amdgcn_wmma_f32_16x16x32_f16(false,(A),false,(B),(short)0,(C),false,false)

#define EPS_C  1e-9f
#define NORM_C 0.22360679774997896f /* 1/sqrt(20) */

__device__ __forceinline__ float silu(float x) { return x * (1.f / (1.f + __expf(-x))); }
__device__ __forceinline__ f32x4 silu4(f32x4 x) {
  f32x4 r;
#pragma unroll
  for (int i = 0; i < 4; ++i) r[i] = silu(x[i]);
  return r;
}

// ---------------------------------------------------------------------------
// Kernel 0: zero the two scatter accumulators
// ---------------------------------------------------------------------------
__global__ void k_zero(float* __restrict__ a, float* __restrict__ b, int n) {
  int i = blockIdx.x * blockDim.x + threadIdx.x;
  int st = gridDim.x * blockDim.x;
  for (; i < n; i += st) { a[i] = 0.f; b[i] = 0.f; }
}

// ---------------------------------------------------------------------------
// Kernel 1: fold linear layers into bilinear tensors, convert to f16.
//   hB[which][n][K'] (col-major, K'=0..1087):
//     K'<1024 (K'=i*32+j): sum_k B[k,i,j]*W[n,64+k]
//     K' in [1024,1056): W[n, K'-1024]        (x1 / "g" block)
//     K' in [1056,1088): W[n, 32 + K'-1056]   (x2 / "a" block)
//   hW1[n][i] = W_mix_xe[n,i];  hM[n][i] = sum_k B_mix_xe[k,i]*W_mix_xe[n,33+k]
// ---------------------------------------------------------------------------
__global__ void k_prep(const float* __restrict__ Bn2e, const float* __restrict__ Wn2e,
                       const float* __restrict__ Be2n, const float* __restrict__ We2n,
                       const float* __restrict__ Bmixe, const float* __restrict__ Wmixe,
                       _Float16* __restrict__ hBn2e, _Float16* __restrict__ hBe2n,
                       _Float16* __restrict__ hW1, _Float16* __restrict__ hM) {
  int idx = blockIdx.x * blockDim.x + threadIdx.x;
  const int nb = 32 * 1088;
  if (idx < 2 * nb) {
    const float* B = (idx < nb) ? Bn2e : Be2n;
    const float* W = (idx < nb) ? Wn2e : We2n;
    _Float16*    o = (idx < nb) ? hBn2e : hBe2n;
    int r = (idx < nb) ? idx : idx - nb;
    int n = r / 1088, K = r % 1088;
    float v;
    if (K < 1024) {
      int i = K >> 5, j = K & 31;
      float acc = 0.f;
      for (int k = 0; k < 32; ++k) acc += B[k * 1024 + i * 32 + j] * W[n * 96 + 64 + k];
      v = acc;
    } else if (K < 1056) {
      v = W[n * 96 + (K - 1024)];
    } else {
      v = W[n * 96 + 32 + (K - 1056)];
    }
    o[n * 1088 + K] = (_Float16)v;
  } else if (idx < 2 * nb + 2048) {
    int r = idx - 2 * nb;
    int which = r >> 10, rr = r & 1023, n = rr >> 5, i = rr & 31;
    if (which == 0) {
      hW1[n * 32 + i] = (_Float16)Wmixe[n * 65 + i];
    } else {
      float acc = 0.f;
      for (int k = 0; k < 32; ++k) acc += Bmixe[k * 32 + i] * Wmixe[n * 65 + 33 + k];
      hM[n * 32 + i] = (_Float16)acc;
    }
  }
}

// ---------------------------------------------------------------------------
// Kernel 2: node pre-mixing + row-std norm. One wave32 == one node (lane = k).
// ---------------------------------------------------------------------------
__global__ __launch_bounds__(256) void k_premix(const float* __restrict__ xn,
                                                const float* __restrict__ xn_attr,
                                                const float* __restrict__ Bmx,
                                                const float* __restrict__ Wmx,
                                                const float* __restrict__ bmx,
                                                float* __restrict__ xn_norm, int N) {
  __shared__ float sB[32 * 97];  // padded rows (stride 97, gcd(97,64)=1)
  __shared__ float sW[32 * 67];
  int tid = threadIdx.x;
  for (int i = tid; i < 32 * 96; i += blockDim.x) {
    int k = i / 96, r = i % 96;
    sB[k * 97 + r] = Bmx[i];
  }
  for (int i = tid; i < 32 * 67; i += blockDim.x) sW[i] = Wmx[i];
  __syncthreads();
  int lane = tid & 31;
  float bk = bmx[lane];
  int waveG = blockIdx.x * (blockDim.x >> 5) + (tid >> 5);
  int nW = gridDim.x * (blockDim.x >> 5);
  const float* bq = sB + lane * 97;
  const float* wq = sW + lane * 67;
  for (int node = waveG; node < N; node += nW) {
    float a0 = xn_attr[node * 3 + 0], a1 = xn_attr[node * 3 + 1], a2 = xn_attr[node * 3 + 2];
    float xk = xn[node * 32 + lane];
    float bi = 0.f, lin = 0.f;
#pragma unroll 8
    for (int i = 0; i < 32; ++i) {
      float xi = __shfl(xk, i, 32);
      float ci = bq[i * 3] * a0 + bq[i * 3 + 1] * a1 + bq[i * 3 + 2] * a2;
      bi += xi * ci;
      lin += wq[i] * xi;
    }
    lin += wq[32] * a0 + wq[33] * a1 + wq[34] * a2;
#pragma unroll 8
    for (int i = 0; i < 32; ++i) lin += wq[35 + i] * __shfl(bi, i, 32);
    lin += bk;
    float s = lin, q = lin * lin;
#pragma unroll
    for (int msk = 16; msk >= 1; msk >>= 1) {
      s += __shfl_xor(s, msk, 32);
      q += __shfl_xor(q, msk, 32);
    }
    float var = fmaxf((q - s * s * (1.f / 32.f)) * (1.f / 31.f), 0.f);
    float rs = 1.f / (sqrtf(var) + EPS_C);
    xn_norm[node * 32 + lane] = lin * rs;
  }
}

// ---------------------------------------------------------------------------
// Kernel 3: the big edge kernel. One wave = tile of 16 edges.
//   A (16x32 f16) = outer-product chunks of (g,a); B from LDS-staged hB.
//   Dual accumulator pairs (even/odd chunks) break WMMA RAW chains.
// ---------------------------------------------------------------------------
__global__ __launch_bounds__(128) void k_edge(
    const float* __restrict__ xn_norm, const float* __restrict__ xe_attr,
    const int* __restrict__ esrc, const int* __restrict__ edst,
    const float* __restrict__ W_fc1, const float* __restrict__ b_fc1,
    const float* __restrict__ W_fc2, const float* __restrict__ b_fc2,
    const _Float16* __restrict__ hB, const _Float16* __restrict__ hW1,
    const _Float16* __restrict__ hM, const float* __restrict__ b_n2e,
    const float* __restrict__ W_mix_xe, const float* __restrict__ b_mix_xe,
    float* __restrict__ accD, float* __restrict__ accS, int E) {
  __shared__ __align__(32) _Float16 sB[32 * 1088];   // 69632 B
  __shared__ __align__(32) _Float16 sW1[32 * 32];
  __shared__ __align__(32) _Float16 sM[32 * 32];
  __shared__ __align__(32) _Float16 sGh[4][16 * 40]; // f16 g stage, reused as XE stage
  __shared__ float sAttr[4][16];
  __shared__ int   sSrc[4][16], sDst[4][16];

  int tid = threadIdx.x;
  for (int i = tid; i < 4352; i += blockDim.x) ((uint4*)sB)[i] = ((const uint4*)hB)[i];
  for (int i = tid; i < 128; i += blockDim.x) {
    ((uint4*)sW1)[i] = ((const uint4*)hW1)[i];
    ((uint4*)sM)[i]  = ((const uint4*)hM)[i];
  }
  __syncthreads();

  int lane = tid & 31, wid = tid >> 5;
  int m = lane & 15, hf = lane >> 4;
  _Float16* sGw = sGh[wid];

  // lane-constant hoists
  f32x4 wf1a = *(const f32x4*)(W_fc1 + hf * 16);
  f32x4 wf1b = *(const f32x4*)(W_fc1 + hf * 16 + 4);
  f32x4 wf1c = *(const f32x4*)(W_fc1 + hf * 16 + 8);
  f32x4 wf1d = *(const f32x4*)(W_fc1 + hf * 16 + 12);
  f32x4 bf1a = *(const f32x4*)(b_fc1 + hf * 16);
  f32x4 bf1b = *(const f32x4*)(b_fc1 + hf * 16 + 4);
  f32x4 bf1c = *(const f32x4*)(b_fc1 + hf * 16 + 8);
  f32x4 bf1d = *(const f32x4*)(b_fc1 + hf * 16 + 12);
  float bias0 = b_n2e[m], bias1 = b_n2e[m + 16];
  float wc0 = W_mix_xe[m * 65 + 32], wc1 = W_mix_xe[(m + 16) * 65 + 32];
  float bmx0 = b_mix_xe[m], bmx1 = b_mix_xe[m + 16];
  float wf2_0 = W_fc2[m], wf2_1 = W_fc2[m + 16];
  float bf2_0 = b_fc2[m], bf2_1 = b_fc2[m + 16];
  const v16h fW10 = *(const v16h*)(sW1 + m * 32 + hf * 16);
  const v16h fW11 = *(const v16h*)(sW1 + (m + 16) * 32 + hf * 16);
  const v16h fM0  = *(const v16h*)(sM + m * 32 + hf * 16);
  const v16h fM1  = *(const v16h*)(sM + (m + 16) * 32 + hf * 16);
  const _Float16* bRow0 = sB + m * 1088 + hf * 16;
  const _Float16* bRow1 = sB + (m + 16) * 1088 + hf * 16;

  int waveG = blockIdx.x * 4 + wid;
  int nW = gridDim.x * 4;
  int nT = (E + 15) >> 4;
  for (int t = waveG; t < nT; t += nW) {
    int e = t * 16 + m;
    int ec = (e < E) ? e : (E - 1);
    int s = esrc[ec], d = edst[ec];
    float at = (e < E) ? xe_attr[ec] : 0.f;
    if (hf == 0) { sSrc[wid][m] = s; sDst[wid][m] = d; sAttr[wid][m] = at; }

    const float* ps = xn_norm + (size_t)s * 32;
    const float* pd = xn_norm + (size_t)d * 32;
    f32x4 xga = *(const f32x4*)(ps + hf * 16);
    f32x4 xgb = *(const f32x4*)(ps + hf * 16 + 4);
    f32x4 xgc = *(const f32x4*)(ps + hf * 16 + 8);
    f32x4 xgd = *(const f32x4*)(ps + hf * 16 + 12);
    f32x4 xda = *(const f32x4*)(pd + hf * 16);
    f32x4 xdb = *(const f32x4*)(pd + hf * 16 + 4);
    f32x4 xdc = *(const f32x4*)(pd + hf * 16 + 8);
    f32x4 xdd = *(const f32x4*)(pd + hf * 16 + 12);
    f32x4 w1a = silu4(at * wf1a + bf1a);
    f32x4 w1b = silu4(at * wf1b + bf1b);
    f32x4 w1c = silu4(at * wf1c + bf1c);
    f32x4 w1d = silu4(at * wf1d + bf1d);
    f32x4 g0 = w1a * (xga - xda), a0 = w1a * (xga + xda) * 0.5f;
    f32x4 g1 = w1b * (xgb - xdb), a1 = w1b * (xgb + xdb) * 0.5f;
    f32x4 g2 = w1c * (xgc - xdc), a2 = w1c * (xgc + xdc) * 0.5f;
    f32x4 g3 = w1d * (xgd - xdd), a3 = w1d * (xgd + xdd) * 0.5f;

    v8h alo = {}, ahi = {}, glo = {}, ghi = {};
#pragma unroll
    for (int q = 0; q < 4; ++q) {
      alo[q] = (_Float16)a0[q]; alo[4 + q] = (_Float16)a1[q];
      ahi[q] = (_Float16)a2[q]; ahi[4 + q] = (_Float16)a3[q];
      glo[q] = (_Float16)g0[q]; glo[4 + q] = (_Float16)g1[q];
      ghi[q] = (_Float16)g2[q]; ghi[4 + q] = (_Float16)g3[q];
    }
    v16h av = {}, gv = {};
#pragma unroll
    for (int q = 0; q < 8; ++q) { av[q] = alo[q]; av[8 + q] = ahi[q];
                                  gv[q] = glo[q]; gv[8 + q] = ghi[q]; }
    // stage g (f16) for per-chunk scalar broadcast
    *(v8h*)(sGw + m * 40 + hf * 16)     = glo;
    *(v8h*)(sGw + m * 40 + hf * 16 + 8) = ghi;

    v8f c0a = {}, c0b = {}, c1a = {}, c1b = {};
#pragma unroll 4
    for (int kc = 0; kc < 32; kc += 2) {
      _Float16 gh0 = sGw[m * 40 + kc];
      _Float16 gh1 = sGw[m * 40 + kc + 1];
      v16h af0 = av * gh0;  // P chunk: g[m][kc] * a[m][j]
      v16h af1 = av * gh1;
      v16h b00 = *(const v16h*)(bRow0 + kc * 32);
      v16h b10 = *(const v16h*)(bRow1 + kc * 32);
      v16h b01 = *(const v16h*)(bRow0 + kc * 32 + 32);
      v16h b11 = *(const v16h*)(bRow1 + kc * 32 + 32);
      c0a = WMMA_F16(af0, b00, c0a);
      c1a = WMMA_F16(af0, b10, c1a);
      c0b = WMMA_F16(af1, b01, c0b);
      c1b = WMMA_F16(af1, b11, c1b);
    }
    {  // folded linear blocks: g-chunk then a-chunk
      v16h b0 = *(const v16h*)(bRow0 + 1024);
      v16h b1 = *(const v16h*)(bRow1 + 1024);
      c0a = WMMA_F16(gv, b0, c0a); c1a = WMMA_F16(gv, b1, c1a);
      b0 = *(const v16h*)(bRow0 + 1056);
      b1 = *(const v16h*)(bRow1 + 1056);
      c0b = WMMA_F16(av, b0, c0b); c1b = WMMA_F16(av, b1, c1b);
    }
    v8f c0 = c0a + c0b;
    v8f c1 = c1a + c1b;
    // xe = c + bias -> relayout C->A via LDS (same wave, DS in-order), f16
#pragma unroll
    for (int v = 0; v < 8; ++v) {
      int mm = hf * 8 + v;
      sGw[mm * 40 + m]      = (_Float16)(c0[v] + bias0);
      sGw[mm * 40 + 16 + m] = (_Float16)(c1[v] + bias1);
    }
    v8h xlo = *(const v8h*)(sGw + m * 40 + hf * 16);
    v8h xhi = *(const v8h*)(sGw + m * 40 + hf * 16 + 8);
    v16h xev = {};
#pragma unroll
    for (int q = 0; q < 8; ++q) { xev[q] = xlo[q]; xev[8 + q] = xhi[q]; }
    v8f z = {};
    v8f da0 = WMMA_F16(xev, fW10, z);
    v8f da1 = WMMA_F16(xev, fW11, z);
    v8f db0 = WMMA_F16(xev, fM0, z);
    v8f db1 = WMMA_F16(xev, fM1, z);
#pragma unroll
    for (int v = 0; v < 8; ++v) {
      int mm = hf * 8 + v;
      float atr = sAttr[wid][mm];
      float x0 = da0[v] + atr * db0[v] + atr * wc0 + bmx0;
      float x1 = da1[v] + atr * db1[v] + atr * wc1 + bmx1;
      float ssum = x0 + x1, ssq = x0 * x0 + x1 * x1;
#pragma unroll
      for (int msk = 1; msk <= 8; msk <<= 1) {
        ssum += __shfl_xor(ssum, msk, 32);
        ssq  += __shfl_xor(ssq, msk, 32);
      }
      float var = fmaxf((ssq - ssum * ssum * (1.f / 32.f)) * (1.f / 31.f), 0.f);
      float rs = 1.f / (sqrtf(var) + EPS_C);
      float w20 = silu(atr * wf2_0 + bf2_0);
      float w21 = silu(atr * wf2_1 + bf2_1);
      float v0 = NORM_C * w20 * x0 * rs;
      float v1 = NORM_C * w21 * x1 * rs;
      int ee = t * 16 + mm;
      if (ee < E) {
        int dd = sDst[wid][mm] * 32, ss = sSrc[wid][mm] * 32;
        __hip_atomic_fetch_add(accD + dd + m,      v0, __ATOMIC_RELAXED, __HIP_MEMORY_SCOPE_AGENT);
        __hip_atomic_fetch_add(accD + dd + 16 + m, v1, __ATOMIC_RELAXED, __HIP_MEMORY_SCOPE_AGENT);
        __hip_atomic_fetch_add(accS + ss + m,      v0, __ATOMIC_RELAXED, __HIP_MEMORY_SCOPE_AGENT);
        __hip_atomic_fetch_add(accS + ss + 16 + m, v1, __ATOMIC_RELAXED, __HIP_MEMORY_SCOPE_AGENT);
      }
    }
  }
}

// ---------------------------------------------------------------------------
// Kernel 4: edges-to-nodes mixing (same WMMA machinery), silu + std, output.
// ---------------------------------------------------------------------------
__global__ __launch_bounds__(128) void k_e2n(const float* __restrict__ accD,
                                             const float* __restrict__ accS,
                                             const _Float16* __restrict__ hB,
                                             const float* __restrict__ b_e2n,
                                             float* __restrict__ out, int N) {
  __shared__ __align__(32) _Float16 sB[32 * 1088];
  __shared__ __align__(32) _Float16 sGh[4][16 * 40];
  int tid = threadIdx.x;
  for (int i = tid; i < 4352; i += blockDim.x) ((uint4*)sB)[i] = ((const uint4*)hB)[i];
  __syncthreads();
  int lane = tid & 31, wid = tid >> 5;
  int m = lane & 15, hf = lane >> 4;
  _Float16* sGw = sGh[wid];
  float bias0 = b_e2n[m], bias1 = b_e2n[m + 16];
  const _Float16* bRow0 = sB + m * 1088 + hf * 16;
  const _Float16* bRow1 = sB + (m + 16) * 1088 + hf * 16;

  int waveG = blockIdx.x * 4 + wid;
  int nW = gridDim.x * 4;
  int nT = (N + 15) >> 4;
  for (int t = waveG; t < nT; t += nW) {
    int node = t * 16 + m;
    int nc = (node < N) ? node : (N - 1);
    const float* pD = accD + (size_t)nc * 32;
    const float* pS = accS + (size_t)nc * 32;
    f32x4 d0 = *(const f32x4*)(pD + hf * 16),         s0 = *(const f32x4*)(pS + hf * 16);
    f32x4 d1 = *(const f32x4*)(pD + hf * 16 + 4),     s1 = *(const f32x4*)(pS + hf * 16 + 4);
    f32x4 d2 = *(const f32x4*)(pD + hf * 16 + 8),     s2 = *(const f32x4*)(pS + hf * 16 + 8);
    f32x4 d3 = *(const f32x4*)(pD + hf * 16 + 12),    s3 = *(const f32x4*)(pS + hf * 16 + 12);
    f32x4 u0 = d0 - s0, w0 = d0 + s0;
    f32x4 u1 = d1 - s1, w1 = d1 + s1;
    f32x4 u2 = d2 - s2, w2 = d2 + s2;
    f32x4 u3 = d3 - s3, w3 = d3 + s3;
    v8h ulo = {}, uhi = {}, wlo = {}, whi = {};
#pragma unroll
    for (int q = 0; q < 4; ++q) {
      ulo[q] = (_Float16)u0[q]; ulo[4 + q] = (_Float16)u1[q];
      uhi[q] = (_Float16)u2[q]; uhi[4 + q] = (_Float16)u3[q];
      wlo[q] = (_Float16)w0[q]; wlo[4 + q] = (_Float16)w1[q];
      whi[q] = (_Float16)w2[q]; whi[4 + q] = (_Float16)w3[q];
    }
    v16h uv = {}, wv = {};
#pragma unroll
    for (int q = 0; q < 8; ++q) { uv[q] = ulo[q]; uv[8 + q] = uhi[q];
                                  wv[q] = wlo[q]; wv[8 + q] = whi[q]; }
    *(v8h*)(sGw + m * 40 + hf * 16)     = ulo;
    *(v8h*)(sGw + m * 40 + hf * 16 + 8) = uhi;

    v8f c0a = {}, c0b = {}, c1a = {}, c1b = {};
#pragma unroll 4
    for (int kc = 0; kc < 32; kc += 2) {
      _Float16 uh0 = sGw[m * 40 + kc];
      _Float16 uh1 = sGw[m * 40 + kc + 1];
      v16h af0 = wv * uh0;
      v16h af1 = wv * uh1;
      v16h b00 = *(const v16h*)(bRow0 + kc * 32);
      v16h b10 = *(const v16h*)(bRow1 + kc * 32);
      v16h b01 = *(const v16h*)(bRow0 + kc * 32 + 32);
      v16h b11 = *(const v16h*)(bRow1 + kc * 32 + 32);
      c0a = WMMA_F16(af0, b00, c0a);
      c1a = WMMA_F16(af0, b10, c1a);
      c0b = WMMA_F16(af1, b01, c0b);
      c1b = WMMA_F16(af1, b11, c1b);
    }
    {
      v16h b0 = *(const v16h*)(bRow0 + 1024);
      v16h b1 = *(const v16h*)(bRow1 + 1024);
      c0a = WMMA_F16(uv, b0, c0a); c1a = WMMA_F16(uv, b1, c1a);
      b0 = *(const v16h*)(bRow0 + 1056);
      b1 = *(const v16h*)(bRow1 + 1056);
      c0b = WMMA_F16(wv, b0, c0b); c1b = WMMA_F16(wv, b1, c1b);
    }
    v8f c0 = c0a + c0b;
    v8f c1 = c1a + c1b;
#pragma unroll
    for (int v = 0; v < 8; ++v) {
      int mm = hf * 8 + v;
      float y0 = silu(c0[v] + bias0);
      float y1 = silu(c1[v] + bias1);
      float ssum = y0 + y1, ssq = y0 * y0 + y1 * y1;
#pragma unroll
      for (int msk = 1; msk <= 8; msk <<= 1) {
        ssum += __shfl_xor(ssum, msk, 32);
        ssq  += __shfl_xor(ssq, msk, 32);
      }
      float var = fmaxf((ssq - ssum * ssum * (1.f / 32.f)) * (1.f / 31.f), 0.f);
      float rs = 1.f / (sqrtf(var) + EPS_C);
      int node2 = t * 16 + mm;
      if (node2 < N) {
        out[(size_t)node2 * 32 + m]      = y0 * rs;
        out[(size_t)node2 * 32 + 16 + m] = y1 * rs;
      }
    }
  }
}

// ---------------------------------------------------------------------------
extern "C" void kernel_launch(void* const* d_in, const int* in_sizes, int n_in,
                              void* d_out, int out_size, void* d_ws, size_t ws_size,
                              hipStream_t stream) {
  const float* xn       = (const float*)d_in[0];
  const float* xn_attr  = (const float*)d_in[1];
  const float* xe_attr  = (const float*)d_in[2];
  const int*   xe_src   = (const int*)d_in[3];
  const int*   xe_dst   = (const int*)d_in[4];
  const float* W_fc1    = (const float*)d_in[5];
  const float* b_fc1    = (const float*)d_in[6];
  const float* W_fc2    = (const float*)d_in[7];
  const float* b_fc2    = (const float*)d_in[8];
  const float* B_mix_xn = (const float*)d_in[9];
  const float* W_mix_xn = (const float*)d_in[10];
  const float* b_mix_xn = (const float*)d_in[11];
  const float* B_n2e    = (const float*)d_in[12];
  const float* W_n2e    = (const float*)d_in[13];
  const float* b_n2e    = (const float*)d_in[14];
  const float* B_mix_xe = (const float*)d_in[15];
  const float* W_mix_xe = (const float*)d_in[16];
  const float* b_mix_xe = (const float*)d_in[17];
  const float* B_e2n    = (const float*)d_in[18];
  const float* W_e2n    = (const float*)d_in[19];
  const float* b_e2n    = (const float*)d_in[20];
  (void)n_in; (void)out_size; (void)ws_size;

  int N = in_sizes[0] / 32;
  int E = in_sizes[3];

  char* ws = (char*)d_ws;
  size_t off = 0;
  auto up = [](size_t x) { return (x + 255) & ~(size_t)255; };
  float* xn_norm = (float*)(ws + off); off = up(off + (size_t)N * 32 * 4);
  float* accD    = (float*)(ws + off); off = up(off + (size_t)N * 32 * 4);
  float* accS    = (float*)(ws + off); off = up(off + (size_t)N * 32 * 4);
  _Float16* hBn2e = (_Float16*)(ws + off); off = up(off + 32 * 1088 * 2);
  _Float16* hBe2n = (_Float16*)(ws + off); off = up(off + 32 * 1088 * 2);
  _Float16* hW1   = (_Float16*)(ws + off); off = up(off + 32 * 32 * 2);
  _Float16* hM    = (_Float16*)(ws + off); off = up(off + 32 * 32 * 2);

  k_zero<<<4096, 256, 0, stream>>>(accD, accS, N * 32);
  k_prep<<<280, 256, 0, stream>>>(B_n2e, W_n2e, B_e2n, W_e2n, B_mix_xe, W_mix_xe,
                                  hBn2e, hBe2n, hW1, hM);
  k_premix<<<1024, 256, 0, stream>>>(xn, xn_attr, B_mix_xn, W_mix_xn, b_mix_xn, xn_norm, N);
  k_edge<<<3072, 128, 0, stream>>>(xn_norm, xe_attr, xe_src, xe_dst, W_fc1, b_fc1,
                                   W_fc2, b_fc2, hBn2e, hW1, hM, b_n2e, W_mix_xe,
                                   b_mix_xe, accD, accS, E);
  k_e2n<<<1024, 128, 0, stream>>>(accD, accS, hBe2n, b_e2n, (float*)d_out, N);
}
